// DynTensorResNet_54460185313347
// MI455X (gfx1250) — matrix operand order repaired and checked by reading, not compiled, not run on hardware
//
#include <hip/hip_runtime.h>
#include <hip/hip_bf16.h>
#include <stdint.h>

// Problem constants (from reference)
#define DIMX   32
#define KR     8
#define CCH    3
#define FDIM   3072          // C*DIM*DIM
#define BATCH  256
#define NLAYER 8
#define NCLASS 10
#define LDT    (NLAYER + 1)  // 9, last-dim stride of X_transformed

typedef __attribute__((ext_vector_type(16))) __bf16       v16bf;
typedef __attribute__((ext_vector_type(8)))  float        v8f;
typedef __attribute__((ext_vector_type(8)))  unsigned int v8u;

// ---------------------------------------------------------------------------
// helpers
// ---------------------------------------------------------------------------
__device__ __forceinline__ unsigned int pack2_bf16(float lo, float hi) {
    // round-to-nearest-even fp32 -> bf16, packed pair (lo in [15:0])
    unsigned int a = __builtin_bit_cast(unsigned int, lo);
    unsigned int b = __builtin_bit_cast(unsigned int, hi);
    a += 0x7FFFu + ((a >> 16) & 1u);
    b += 0x7FFFu + ((b >> 16) & 1u);
    return (a >> 16) | (b & 0xFFFF0000u);
}

// parallel Gauss-Jordan inverse of n x n (augmented n x 2n, row-major, pitch=2n)
// Matrices here are SPD / diagonally dominant, so no pivoting needed.
__device__ void gauss_jordan(float* aug, int n, int tid, int nth) {
    const int pitch = 2 * n;
    for (int k = 0; k < n; ++k) {
        __syncthreads();
        float piv = aug[k * pitch + k];
        __syncthreads();
        float inv = 1.0f / piv;
        for (int c = tid; c < pitch; c += nth) aug[k * pitch + c] *= inv;
        __syncthreads();
        int tpr = nth / n;                 // threads per row
        int r   = tid / tpr;
        float f = (r < n) ? aug[r * pitch + k] : 0.0f;
        __syncthreads();
        if (r < n && r != k) {
            for (int c = tid % tpr; c < pitch; c += tpr)
                aug[r * pitch + c] -= f * aug[k * pitch + c];
        }
    }
    __syncthreads();
}

__device__ __forceinline__ void mm3(const float* A, const float* B, float* Cm) {
    #pragma unroll
    for (int i = 0; i < 3; ++i)
        #pragma unroll
        for (int j = 0; j < 3; ++j) {
            float s = 0.f;
            #pragma unroll
            for (int k = 0; k < 3; ++k) s += A[i * 3 + k] * B[k * 3 + j];
            Cm[i * 3 + j] = s;
        }
}

__device__ __forceinline__ void inv3(const float* A, float* R) {
    float a = A[0], b = A[1], c = A[2];
    float d = A[3], e = A[4], f = A[5];
    float g = A[6], h = A[7], i = A[8];
    float det = a * (e * i - f * h) - b * (d * i - f * g) + c * (d * h - e * g);
    float id = 1.0f / det;
    R[0] = (e * i - f * h) * id; R[1] = (c * h - b * i) * id; R[2] = (b * f - c * e) * id;
    R[3] = (f * g - d * i) * id; R[4] = (a * i - c * g) * id; R[5] = (c * d - a * f) * id;
    R[6] = (d * h - e * g) * id; R[7] = (b * g - a * h) * id; R[8] = (a * e - b * d) * id;
}

// ---------------------------------------------------------------------------
// WMMA GEMM:  out[m,n] = act( sum_k A[m,k] * W[n,k] + bias[n] )
// A: M x K fp32 row-major, W: N x K fp32 row-major.
// Block tile 128(M) x 64(N), 8 waves (wave32), each wave 32x32 via 2x2 WMMA
// fragments, K-chunks of 32 with fp32->bf16 conversion while staging to LDS.
// ---------------------------------------------------------------------------
__global__ __launch_bounds__(256) void gemm_bias_act_wmma(
    const float* __restrict__ A, const float* __restrict__ W,
    const float* __restrict__ bias,
    float* __restrict__ outC, float* __restrict__ outS, int slot,
    int M, int N, int K, int relu)
{
    __shared__ uint32_t As[128][17];   // [m][k-pair], +1 pad vs 16 to dodge bank conflicts
    __shared__ uint32_t Bs[64][17];    // [n][k-pair]

    const int tid  = threadIdx.x;
    const int lane = tid & 31;
    const int wave = tid >> 5;
    const int half = lane >> 4;        // 0: lanes 0-15, 1: lanes 16-31
    const int la   = lane & 15;

    const int nbm = M >> 7;
    const int m0  = (blockIdx.x % nbm) << 7;
    const int n0  = (blockIdx.x / nbm) << 6;

    const int wm = (wave & 3) << 5;    // wave M offset: 0,32,64,96
    const int wn = (wave >> 2) << 5;   // wave N offset: 0,32

    const v8f vzero = {0.f, 0.f, 0.f, 0.f, 0.f, 0.f, 0.f, 0.f};
    v8f acc[2][2];
    acc[0][0] = vzero; acc[0][1] = vzero; acc[1][0] = vzero; acc[1][1] = vzero;

    // cooperative tile loads: A 128x32 (16 elem/thread), B 64x32 (8 elem/thread)
    const int arow = tid >> 1;              // 0..127
    const int akp  = (tid & 1) << 3;        // pair offset 0 or 8
    const int brow = tid >> 2;              // 0..63
    const int bkp  = (tid & 3) << 2;        // pair offset 0,4,8,12

    const float* Abase = A + (size_t)(m0 + arow) * K + (akp << 1);
    const float* Wbase = W + (size_t)(n0 + brow) * K + (bkp << 1);

    const int nk = K >> 5;
    for (int kt = 0; kt < nk; ++kt) {
        const float2* ap = (const float2*)(Abase + kt * 32);
        #pragma unroll
        for (int i = 0; i < 8; ++i) {
            float2 f = ap[i];
            As[arow][akp + i] = pack2_bf16(f.x, f.y);
        }
        const float2* bp = (const float2*)(Wbase + kt * 32);
        #pragma unroll
        for (int i = 0; i < 4; ++i) {
            float2 f = bp[i];
            Bs[brow][bkp + i] = pack2_bf16(f.x, f.y);
        }
        if (kt + 1 < nk) {                 // global_prefetch_b8 of next K tile
            __builtin_prefetch(Abase + (kt + 1) * 32, 0, 1);
            __builtin_prefetch(Wbase + (kt + 1) * 32, 0, 1);
        }
        __syncthreads();

        // A fragment, 16-bit 16x32 layout: lanes<16 K=0..7/16..23, lanes>=16 K=8..15/24..31
        v8u afr[2], bfr[2];
        #pragma unroll
        for (int mi = 0; mi < 2; ++mi) {
            int m  = wm + (mi << 4) + la;
            int kb = half << 2;
            #pragma unroll
            for (int v = 0; v < 4; ++v) {
                afr[mi][v]     = As[m][kb + v];
                afr[mi][4 + v] = As[m][8 + kb + v];
            }
        }
        // B fragment, 16-bit 32x16 layout: lanes<16 K=0..7/16..23, lanes>=16 K=8..15/24..31
        #pragma unroll
        for (int ni = 0; ni < 2; ++ni) {
            int n = wn + (ni << 4) + la;
            #pragma unroll
            for (int v = 0; v < 8; ++v)
                bfr[ni][v] = Bs[n][(half << 2) + ((v >> 2) << 3) + (v & 3)];
        }
        #pragma unroll
        for (int mi = 0; mi < 2; ++mi)
            #pragma unroll
            for (int ni = 0; ni < 2; ++ni)
                acc[mi][ni] = __builtin_amdgcn_wmma_f32_16x16x32_bf16(
                    false, __builtin_bit_cast(v16bf, afr[mi]),
                    false, __builtin_bit_cast(v16bf, bfr[ni]),
                    (short)0, acc[mi][ni], false, false);
        __syncthreads();
    }

    // epilogue: C/D layout = VGPR r, lanes<16 -> (M=r, N=lane), lanes>=16 -> (M=8+r)
    #pragma unroll
    for (int mi = 0; mi < 2; ++mi) {
        #pragma unroll
        for (int ni = 0; ni < 2; ++ni) {
            int col  = n0 + wn + (ni << 4) + la;
            float bv = bias ? bias[col] : 0.0f;
            #pragma unroll
            for (int r = 0; r < 8; ++r) {
                int row = m0 + wm + (mi << 4) + (half << 3) + r;
                float v = acc[mi][ni][r] + bv;
                if (relu) v = fmaxf(v, 0.0f);
                outC[(size_t)row * N + col] = v;
                if (outS) outS[((size_t)row * N + col) * LDT + slot] = v;
            }
        }
    }
}

// ---------------------------------------------------------------------------
// prep: slice S,U1,U2,U3 from X, Tucker-reconstruct T0, store factor state
// one block (128 threads) per sample
// ---------------------------------------------------------------------------
__global__ __launch_bounds__(128) void prep_kernel(
    const float* __restrict__ X, float* __restrict__ T0, float* __restrict__ SU)
{
    const int b   = blockIdx.x;
    const int tid = threadIdx.x;
    __shared__ float sS[3][8][8];
    __shared__ float sU1[3][3];
    __shared__ float sU2[32][8];
    __shared__ float sU3[32][8];
    __shared__ float sT1[3][8][32];
    __shared__ float sT2[3][32][32];
    const float* Xb = X + (size_t)b * 6 * 256;   // (6, 32, 8) per channel row-major

    for (int idx = tid; idx < 192; idx += 128) {
        int a = idx >> 6, q = (idx >> 3) & 7, r = idx & 7;
        sS[a][q][r] = Xb[a * 256 + q * 8 + r];
    }
    if (tid < 9) sU1[tid / 3][tid % 3] = Xb[3 * 256 + (tid / 3) * 8 + (tid % 3)];
    for (int idx = tid; idx < 256; idx += 128) sU2[idx >> 3][idx & 7] = Xb[4 * 256 + idx];
    for (int idx = tid; idx < 256; idx += 128) sU3[idx >> 3][idx & 7] = Xb[5 * 256 + idx];
    __syncthreads();

    for (int idx = tid; idx < 768; idx += 128) {          // T1[p,q,j] = S . U3
        int p = idx >> 8, q = (idx >> 5) & 7, j = idx & 31;
        float s = 0.f;
        #pragma unroll
        for (int r = 0; r < 8; ++r) s += sS[p][q][r] * sU3[j][r];
        sT1[p][q][j] = s;
    }
    __syncthreads();
    for (int idx = tid; idx < 3072; idx += 128) {         // T2[p,i,j] = T1 . U2
        int p = idx >> 10, i = (idx >> 5) & 31, j = idx & 31;
        float s = 0.f;
        #pragma unroll
        for (int q = 0; q < 8; ++q) s += sT1[p][q][j] * sU2[i][q];
        sT2[p][i][j] = s;
    }
    __syncthreads();
    for (int idx = tid; idx < 3072; idx += 128) {         // out[a,i,j] = U1 . T2
        int a = idx >> 10, i = (idx >> 5) & 31, j = idx & 31;
        float s = 0.f;
        #pragma unroll
        for (int p = 0; p < 3; ++p) s += sU1[a][p] * sT2[p][i][j];
        T0[(size_t)b * FDIM + idx] = s;
    }
    float* su = SU + (size_t)b * 768;
    for (int idx = tid; idx < 192; idx += 128) su[idx] = (&sS[0][0][0])[idx];
    if (tid < 9) su[192 + tid] = (&sU1[0][0])[tid];
    for (int idx = tid; idx < 256; idx += 128) su[256 + idx] = (&sU2[0][0])[idx];
    for (int idx = tid; idx < 256; idx += 128) su[512 + idx] = (&sU3[0][0])[idx];
}

// ---------------------------------------------------------------------------
// per-sample tensor/retraction update: one block (256 threads) per sample
// ---------------------------------------------------------------------------
__global__ __launch_bounds__(256) void update_kernel(
    const float* __restrict__ dY, float* __restrict__ SU,
    float* __restrict__ Xc, float* __restrict__ outS, int slot, float h)
{
    const int b   = blockIdx.x;
    const int tid = threadIdx.x;

    __shared__ float big[6400];          // aliased pool: dY|A1 -> Cayley temps -> Tucker temps
    __shared__ float sS[3][8][8];
    __shared__ float sdS[3][8][8];
    __shared__ float sU1[3][3];
    __shared__ float sU2[32][8];
    __shared__ float sU3[32][8];
    __shared__ float sB1[3][8][32];
    __shared__ float sA2[3][8][32];      // == Y3[p,q,j]
    __shared__ float sY1[3][8][8];
    __shared__ float sY2u[32][24];
    __shared__ float sY3u[32][24];
    __shared__ float sS1u[8][24];
    __shared__ float sS2u[8][24];
    __shared__ float sP[32][8];
    __shared__ float sdU2[32][8];
    __shared__ float sdU3[32][8];
    __shared__ float sFU2[32][8];
    __shared__ float sFU3[32][8];
    __shared__ float sT8[8][8];
    __shared__ float sAug8[8][16];
    __shared__ float sU1n[3][3];
    __shared__ float sU2n[32][8];
    __shared__ float sU3n[32][8];
    __shared__ float sSm[32];            // [0..8]=G1, [9..17]=P1

    float* dYs = big;                    // 3072 (dies after A1/B1)
    float* A1  = big + 3072;             // 3072 (dies after A2/Y2u)

    const float* dYb = dY + (size_t)b * FDIM;
    float* su = SU + (size_t)b * 768;

    for (int idx = tid; idx < 3072; idx += 256) dYs[idx] = dYb[idx];
    for (int idx = tid; idx < 192; idx += 256) (&sS[0][0][0])[idx] = su[idx];
    if (tid < 9) (&sU1[0][0])[tid] = su[192 + tid];
    (&sU2[0][0])[tid] = su[256 + tid];
    (&sU3[0][0])[tid] = su[512 + tid];
    __syncthreads();

    // --- unfoldings of S, A1 = dY x U1, B1 = dY x U2 -----------------------
    for (int idx = tid; idx < 192; idx += 256) {
        int p = idx >> 6, q = (idx >> 3) & 7, r = idx & 7;
        float v = sS[p][q][r];
        sS1u[q][p * 8 + r] = v;          // mode-2 unfolding
        sS2u[r][p * 8 + q] = v;          // mode-3 unfolding
    }
    for (int idx = tid; idx < 3072; idx += 256) {
        int p = idx >> 10, i = (idx >> 5) & 31, j = idx & 31;
        float s = 0.f;
        #pragma unroll
        for (int a = 0; a < 3; ++a) s += sU1[a][p] * dYs[a * 1024 + i * 32 + j];
        A1[idx] = s;
    }
    for (int idx = tid; idx < 768; idx += 256) {
        int a = idx >> 8, q = (idx >> 5) & 7, j = idx & 31;
        float s = 0.f;
        #pragma unroll
        for (int i = 0; i < 32; ++i) s += dYs[a * 1024 + i * 32 + j] * sU2[i][q];
        sB1[a][q][j] = s;
    }
    __syncthreads();

    // --- A2 (=Y3), Y2u, Y1 -------------------------------------------------
    for (int idx = tid; idx < 768; idx += 256) {
        int p = idx >> 8, q = (idx >> 5) & 7, j = idx & 31;
        float s = 0.f;
        #pragma unroll
        for (int i = 0; i < 32; ++i) s += A1[p * 1024 + i * 32 + j] * sU2[i][q];
        sA2[p][q][j] = s;
    }
    for (int idx = tid; idx < 768; idx += 256) {
        int i = idx / 24, p = (idx % 24) >> 3, r = idx & 7;
        float s = 0.f;
        #pragma unroll
        for (int j = 0; j < 32; ++j) s += A1[p * 1024 + i * 32 + j] * sU3[j][r];
        sY2u[i][p * 8 + r] = s;
    }
    for (int idx = tid; idx < 192; idx += 256) {
        int a = idx >> 6, q = (idx >> 3) & 7, r = idx & 7;
        float s = 0.f;
        #pragma unroll
        for (int j = 0; j < 32; ++j) s += sB1[a][q][j] * sU3[j][r];
        sY1[a][q][r] = s;
    }
    __syncthreads();

    // --- dS, Y3u, Gram matrices -------------------------------------------
    for (int idx = tid; idx < 192; idx += 256) {
        int p = idx >> 6, q = (idx >> 3) & 7, r = idx & 7;
        float s = 0.f;
        #pragma unroll
        for (int j = 0; j < 32; ++j) s += sA2[p][q][j] * sU3[j][r];
        sdS[p][q][r] = s;
    }
    for (int idx = tid; idx < 768; idx += 256) {
        int j = idx / 24, p = (idx % 24) >> 3, q = idx & 7;
        sY3u[j][p * 8 + q] = sA2[p][q][j];
    }
    if (tid < 18) {                      // G1 = Sf Sf^T, P1 = Y1f Sf^T  (3x3 each)
        int x = (tid % 9) / 3, y = (tid % 9) % 3;
        const float* L = (tid < 9) ? &sS[0][0][0] : &sY1[0][0][0];
        float s = 0.f;
        for (int c = 0; c < 64; ++c) s += L[x * 64 + c] * (&sS[0][0][0])[y * 64 + c];
        sSm[tid] = s;
    }
    // G2 = S1u S1u^T (8x8) into augmented form, P2 = Y2u S1u^T (32x8)
    for (int idx = tid; idx < 64; idx += 256) {
        int x = idx >> 3, y = idx & 7;
        float s = 0.f;
        #pragma unroll
        for (int c = 0; c < 24; ++c) s += sS1u[x][c] * sS1u[y][c];
        sAug8[x][y] = s;
        sAug8[x][8 + y] = (x == y) ? 1.0f : 0.0f;
    }
    {
        int i = tid >> 3, y = tid & 7;
        float s = 0.f;
        #pragma unroll
        for (int c = 0; c < 24; ++c) s += sY2u[i][c] * sS1u[y][c];
        sP[i][y] = s;
    }
    gauss_jordan(&sAug8[0][0], 8, tid, 256);
    {
        int i = tid >> 3, y = tid & 7;   // dU2 = P2 * G2inv
        float s = 0.f;
        #pragma unroll
        for (int z = 0; z < 8; ++z) s += sP[i][z] * sAug8[z][8 + y];
        sdU2[i][y] = s;
    }
    __syncthreads();

    // --- G3 / P3 / dU3 ------------------------------------------------------
    for (int idx = tid; idx < 64; idx += 256) {
        int x = idx >> 3, y = idx & 7;
        float s = 0.f;
        #pragma unroll
        for (int c = 0; c < 24; ++c) s += sS2u[x][c] * sS2u[y][c];
        sAug8[x][y] = s;
        sAug8[x][8 + y] = (x == y) ? 1.0f : 0.0f;
    }
    {
        int i = tid >> 3, y = tid & 7;
        float s = 0.f;
        #pragma unroll
        for (int c = 0; c < 24; ++c) s += sY3u[i][c] * sS2u[y][c];
        sP[i][y] = s;
    }
    gauss_jordan(&sAug8[0][0], 8, tid, 256);
    {
        int i = tid >> 3, y = tid & 7;
        float s = 0.f;
        #pragma unroll
        for (int z = 0; z < 8; ++z) s += sP[i][z] * sAug8[z][8 + y];
        sdU3[i][y] = s;
    }
    __syncthreads();

    // --- thread 0: dU1, FU1, Cayley(U1) all 3x3 -----------------------------
    if (tid == 0) {
        float G1i[9], dU1[9], UTdU[9], t9[9], FU1[9], Cm1[9], Dm1[9], DT1[9];
        float E1[9], E1i[9], t2[9], cay1[9];
        inv3(&sSm[0], G1i);
        mm3(&sSm[9], G1i, dU1);                         // dU1 = P1 G1^-1
        #pragma unroll
        for (int i = 0; i < 3; ++i)
            for (int j = 0; j < 3; ++j) {
                float s = 0.f;
                for (int k = 0; k < 3; ++k) s += sU1[k][i] * dU1[k * 3 + j];
                UTdU[i * 3 + j] = s;
            }
        mm3(&sU1[0][0], UTdU, t9);
        #pragma unroll
        for (int i = 0; i < 9; ++i) FU1[i] = dU1[i] - t9[i];
        #pragma unroll
        for (int i = 0; i < 3; ++i)
            for (int j = 0; j < 3; ++j) {
                float s1 = 0.f, s2 = 0.f;
                for (int k = 0; k < 3; ++k) { s1 += FU1[i * 3 + k] * sU1[j][k];
                                              s2 += sU1[i][k] * FU1[j * 3 + k]; }
                Cm1[i * 3 + j] = s1;                    // FU1 U1^T
                Dm1[i * 3 + j] = -h * s2;               // -h U1 FU1^T
            }
        #pragma unroll
        for (int i = 0; i < 3; ++i)
            for (int j = 0; j < 3; ++j) DT1[i * 3 + j] = Dm1[j * 3 + i];
        mm3(DT1, Cm1, t2);
        #pragma unroll
        for (int i = 0; i < 9; ++i) E1[i] = ((i % 4 == 0) ? 1.0f : 0.0f) - 0.5f * t2[i];
        inv3(E1, E1i);
        mm3(Cm1, E1i, t2);
        mm3(t2, DT1, t9);
        #pragma unroll
        for (int i = 0; i < 9; ++i) cay1[i] = ((i % 4 == 0) ? 1.0f : 0.0f) + t9[i];
        #pragma unroll
        for (int i = 0; i < 3; ++i)
            for (int j = 0; j < 3; ++j) {
                float s = 0.f;
                for (int k = 0; k < 3; ++k) s += cay1[i * 3 + k] * sU1[k][j];
                sU1n[i][j] = s;
            }
    }

    // --- FU2 / FU3 (tangent projection) -------------------------------------
    for (int idx = tid; idx < 64; idx += 256) {          // T8 = U2^T dU2
        int q = idx >> 3, y = idx & 7;
        float s = 0.f;
        #pragma unroll
        for (int i = 0; i < 32; ++i) s += sU2[i][q] * sdU2[i][y];
        sT8[q][y] = s;
    }
    __syncthreads();
    {
        int i = tid >> 3, y = tid & 7;
        float s = sdU2[i][y];
        #pragma unroll
        for (int q = 0; q < 8; ++q) s -= sU2[i][q] * sT8[q][y];
        sFU2[i][y] = s;
    }
    __syncthreads();
    for (int idx = tid; idx < 64; idx += 256) {          // T8 = U3^T dU3
        int q = idx >> 3, y = idx & 7;
        float s = 0.f;
        #pragma unroll
        for (int i = 0; i < 32; ++i) s += sU3[i][q] * sdU3[i][y];
        sT8[q][y] = s;
    }
    __syncthreads();
    {
        int i = tid >> 3, y = tid & 7;
        float s = sdU3[i][y];
        #pragma unroll
        for (int q = 0; q < 8; ++q) s -= sU3[i][q] * sT8[q][y];
        sFU3[i][y] = s;
    }
    __syncthreads();

    // --- Cayley retraction for U2 and U3 (32x32, pool) ----------------------
    // Cm = h*FU*U^T, DT = -Cm  =>  E = I + 0.5*Cm^2, cay = I - Cm E^-1 Cm
    float* Cm  = big;            // 1024
    float* M2  = big + 1024;     // 1024
    float* Aug = big + 2048;     // 2048 (32 x 64)
    float* R1  = big + 4096;     // 1024
    float* cy  = big + 5120;     // 1024
    for (int pass = 0; pass < 2; ++pass) {
        const float (*FU)[8] = (pass == 0) ? sFU2 : sFU3;
        const float (*Uo)[8] = (pass == 0) ? sU2  : sU3;
        float (*Un)[8]       = (pass == 0) ? sU2n : sU3n;

        for (int idx = tid; idx < 1024; idx += 256) {
            int i = idx >> 5, j = idx & 31;
            float s = 0.f;
            #pragma unroll
            for (int q = 0; q < 8; ++q) s += FU[i][q] * Uo[j][q];
            Cm[idx] = h * s;
        }
        __syncthreads();
        for (int idx = tid; idx < 1024; idx += 256) {
            int i = idx >> 5, j = idx & 31;
            float s = 0.f;
            #pragma unroll
            for (int k = 0; k < 32; ++k) s += Cm[i * 32 + k] * Cm[k * 32 + j];
            M2[idx] = s;
        }
        __syncthreads();
        for (int idx = tid; idx < 2048; idx += 256) {    // [E | I]
            int r = idx >> 6, c = idx & 63;
            float v;
            if (c < 32) v = ((r == c) ? 1.0f : 0.0f) + 0.5f * M2[r * 32 + c];
            else        v = ((c - 32) == r) ? 1.0f : 0.0f;
            Aug[idx] = v;
        }
        gauss_jordan(Aug, 32, tid, 256);
        for (int idx = tid; idx < 1024; idx += 256) {    // R1 = Cm * E^-1
            int i = idx >> 5, j = idx & 31;
            float s = 0.f;
            #pragma unroll
            for (int k = 0; k < 32; ++k) s += Cm[i * 32 + k] * Aug[k * 64 + 32 + j];
            R1[idx] = s;
        }
        __syncthreads();
        for (int idx = tid; idx < 1024; idx += 256) {    // cay = I - R1 * Cm
            int i = idx >> 5, j = idx & 31;
            float s = (i == j) ? 1.0f : 0.0f;
            #pragma unroll
            for (int k = 0; k < 32; ++k) s -= R1[i * 32 + k] * Cm[k * 32 + j];
            cy[idx] = s;
        }
        __syncthreads();
        {
            int i = tid >> 3, q = tid & 7;               // Un = cay * U
            float s = 0.f;
            #pragma unroll
            for (int k = 0; k < 32; ++k) s += cy[i * 32 + k] * Uo[k][q];
            Un[i][q] = s;
        }
        __syncthreads();
    }

    // --- Sn = S + h*dS ------------------------------------------------------
    for (int idx = tid; idx < 192; idx += 256)
        (&sS[0][0][0])[idx] += h * (&sdS[0][0][0])[idx];
    __syncthreads();

    // --- Xn = tucker(Sn, U1n, U2n, U3n) -------------------------------------
    float* T1 = big;             // 768
    float* T2 = big + 1024;      // 3072
    for (int idx = tid; idx < 768; idx += 256) {
        int p = idx >> 8, q = (idx >> 5) & 7, j = idx & 31;
        float s = 0.f;
        #pragma unroll
        for (int r = 0; r < 8; ++r) s += sS[p][q][r] * sU3n[j][r];
        T1[idx] = s;
    }
    __syncthreads();
    for (int idx = tid; idx < 3072; idx += 256) {
        int p = idx >> 10, i = (idx >> 5) & 31, j = idx & 31;
        float s = 0.f;
        #pragma unroll
        for (int q = 0; q < 8; ++q) s += T1[p * 256 + q * 32 + j] * sU2n[i][q];
        T2[idx] = s;
    }
    __syncthreads();
    for (int idx = tid; idx < 3072; idx += 256) {
        int a = idx >> 10, i = (idx >> 5) & 31, j = idx & 31;
        float s = 0.f;
        #pragma unroll
        for (int p = 0; p < 3; ++p) s += sU1n[a][p] * T2[p * 1024 + i * 32 + j];
        Xc[(size_t)b * FDIM + idx] = s;
        outS[((size_t)b * FDIM + idx) * LDT + slot] = s;
    }

    // --- persist updated factor state ---------------------------------------
    for (int idx = tid; idx < 192; idx += 256) su[idx] = (&sS[0][0][0])[idx];
    if (tid < 9) su[192 + tid] = (&sU1n[0][0])[tid];
    su[256 + tid] = (&sU2n[0][0])[tid];
    su[512 + tid] = (&sU3n[0][0])[tid];
}

// ---------------------------------------------------------------------------
// classifier + softmax: one block (256 threads) per sample
// ---------------------------------------------------------------------------
__global__ __launch_bounds__(256) void classifier_kernel(
    const float* __restrict__ Xf, const float* __restrict__ Wc,
    const float* __restrict__ bc, float* __restrict__ pred,
    float* __restrict__ logits)
{
    const int b   = blockIdx.x;
    const int tid = threadIdx.x;
    __shared__ float red[NCLASS][256];
    float acc[NCLASS];
    #pragma unroll
    for (int c = 0; c < NCLASS; ++c) acc[c] = 0.f;
    for (int f = tid; f < FDIM; f += 256) {
        float x = Xf[(size_t)b * FDIM + f];
        #pragma unroll
        for (int c = 0; c < NCLASS; ++c) acc[c] += x * Wc[c * FDIM + f];
    }
    #pragma unroll
    for (int c = 0; c < NCLASS; ++c) red[c][tid] = acc[c];
    __syncthreads();
    for (int s = 128; s > 0; s >>= 1) {
        if (tid < s)
            #pragma unroll
            for (int c = 0; c < NCLASS; ++c) red[c][tid] += red[c][tid + s];
        __syncthreads();
    }
    if (tid == 0) {
        float l[NCLASS], e[NCLASS], m = -3.0e38f, sum = 0.f;
        #pragma unroll
        for (int c = 0; c < NCLASS; ++c) { l[c] = red[c][0] + bc[c]; m = fmaxf(m, l[c]); }
        #pragma unroll
        for (int c = 0; c < NCLASS; ++c) { e[c] = __expf(l[c] - m); sum += e[c]; }
        float inv = 1.0f / sum;
        #pragma unroll
        for (int c = 0; c < NCLASS; ++c) {
            logits[b * NCLASS + c] = l[c];
            pred[b * NCLASS + c]   = e[c] * inv;
        }
    }
}

// ---------------------------------------------------------------------------
// launcher
// ---------------------------------------------------------------------------
extern "C" void kernel_launch(void* const* d_in, const int* in_sizes, int n_in,
                              void* d_out, int out_size, void* d_ws, size_t ws_size,
                              hipStream_t stream) {
    const float* X  = (const float*)d_in[0];   // (256, 6, 256)
    const float* W0 = (const float*)d_in[1];   // (3072, 3072)
    const float* Ws = (const float*)d_in[2];   // (8, 3072, 3072)
    const float* bs = (const float*)d_in[3];   // (8, 3072)
    const float* Wc = (const float*)d_in[4];   // (10, 3072)
    const float* bc = (const float*)d_in[5];   // (10,)

    float* out_pred   = (float*)d_out;                         // (256,10)
    float* out_logit  = out_pred  + (size_t)BATCH * NCLASS;    // (256,10)
    float* out_xtrans = out_logit + (size_t)BATCH * NCLASS;    // (256,3072,9)

    float* ws = (float*)d_ws;
    float* Xc = ws;                                            // 786432 f
    float* dY = ws + (size_t)BATCH * FDIM;                     // 786432 f (also T0)
    float* SU = ws + 2 * (size_t)BATCH * FDIM;                 // 196608 f

    const int gemm_blocks = (BATCH / 128) * (FDIM / 64);       // 96
    const float h = 1.0f / (float)NLAYER;

    // Tucker reconstruction of initial state -> T0 (in dY buffer)
    prep_kernel<<<BATCH, 128, 0, stream>>>(X, dY, SU);

    // X0 = T0 @ W0^T   (no bias / no relu), also X_transformed[:,:,0]
    gemm_bias_act_wmma<<<gemm_blocks, 256, 0, stream>>>(
        dY, W0, nullptr, Xc, out_xtrans, 0, BATCH, FDIM, FDIM, 0);

    for (int l = 0; l < NLAYER; ++l) {
        // dY = relu(Xc @ W_l^T + b_l)
        gemm_bias_act_wmma<<<gemm_blocks, 256, 0, stream>>>(
            Xc, Ws + (size_t)l * FDIM * FDIM, bs + (size_t)l * FDIM,
            dY, nullptr, 0, BATCH, FDIM, FDIM, 1);
        // dynamic low-rank update + Tucker rebuild -> Xc, X_transformed[:,:,l+1]
        update_kernel<<<BATCH, 256, 0, stream>>>(dY, SU, Xc, out_xtrans, l + 1, h);
    }

    // logits + softmax
    classifier_kernel<<<BATCH, 256, 0, stream>>>(Xc, Wc, bc, out_pred, out_logit);
}
